// GIN_87711822119196
// MI455X (gfx1250) — matrix operand order repaired and checked by reading, not compiled, not run on hardware
//
#include <hip/hip_runtime.h>

// ---------------------------------------------------------------------------
// GIN forward on gfx1250 (MI455X): scatter-add aggregation + fused WMMA MLP.
// GEMMs run on v_wmma_f32_16x16x32_bf16 with bf16 hi/lo split (3 products)
// for ~fp32 accuracy at f32 accumulation. Workload is L2-bandwidth bound
// (node matrix = 25.6MB << 192MB L2), so WMMA throughput is ample.
// ---------------------------------------------------------------------------

typedef __attribute__((ext_vector_type(16))) __bf16 bf16x16;
typedef __attribute__((ext_vector_type(8)))  __bf16 bf16x8;
typedef __attribute__((ext_vector_type(8)))  float  f32x8;

#define HID   128
#define TM    64           // nodes per block in the MLP kernel
#define NMAT  8            // total 128x128 weight matrices (2 per layer x 4)
#define MATSZ (HID*HID)    // 16384 elements per matrix

static __device__ __forceinline__ f32x8 wmma_bf16(bf16x16 a, bf16x16 b, f32x8 c) {
    return __builtin_amdgcn_wmma_f32_16x16x32_bf16(
        /*neg_a=*/false, a, /*neg_b=*/false, b,
        /*c_mod=*/(short)0, c, /*reuse_a=*/false, /*reuse_b=*/false);
}

// A-matrix (16x32 bf16) per-lane fragment: two contiguous 8-element (16B) runs.
// laneLo: K {kb..kb+15} with kb = kt*32
// laneHi: same shifted by +16.
static __device__ __forceinline__ bf16x16 load_a(const __bf16* rowbase, int kb) {
    union { bf16x16 v; bf16x8 h[2]; } u;
    u.h[0] = *(const bf16x8*)(rowbase + kb);
    u.h[1] = *(const bf16x8*)(rowbase + kb + 8);
    return u.v;
}

// ---------------------------------------------------------------------------
// Weight packing: fp32 [n][k] -> bf16 hi/lo row-major (split-fp32 emulation)
// ---------------------------------------------------------------------------
__global__ __launch_bounds__(256) void pack_kernel(
    const float* __restrict__ W1_0, const float* __restrict__ W2_0,
    const float* __restrict__ W1s,  const float* __restrict__ W2s,
    __bf16* __restrict__ whi, __bf16* __restrict__ wlo)
{
    int mid = blockIdx.y;                 // 0..7: W1_0,W2_0,W1s0,W2s0,W1s1,...
    const float* W;
    if (mid == 0)      W = W1_0;
    else if (mid == 1) W = W2_0;
    else {
        int l = (mid - 2) >> 1;
        W = (((mid & 1) == 0) ? W1s : W2s) + (size_t)l * MATSZ;
    }
    int f = blockIdx.x * 256 + threadIdx.x;   // 0..16383
    float w = W[f];
    __bf16 h = (__bf16)w;
    whi[(size_t)mid * MATSZ + f] = h;
    wlo[(size_t)mid * MATSZ + f] = (__bf16)(w - (float)h);
}

// ---------------------------------------------------------------------------
// Fold bias + BN(eval) into per-channel scale/shift:  y = acc*s + t
//   s = g * rsqrt(v+eps);  t = (b1 - m)*s + be
// ---------------------------------------------------------------------------
__global__ void bnprep_kernel(
    const float* __restrict__ g0, const float* __restrict__ be0,
    const float* __restrict__ m0, const float* __restrict__ v0,
    const float* __restrict__ b1_0,
    const float* __restrict__ gs, const float* __restrict__ bes,
    const float* __restrict__ ms, const float* __restrict__ vs,
    const float* __restrict__ b1s,
    float* __restrict__ bns, float* __restrict__ bnt)
{
    int tid = threadIdx.x;               // 512 threads: 4 layers x 128
    int l = tid >> 7, n = tid & 127;
    float g, be, m, v, b1;
    if (l == 0) { g = g0[n]; be = be0[n]; m = m0[n]; v = v0[n]; b1 = b1_0[n]; }
    else {
        int i = l - 1;
        g  = gs [i*HID + n]; be = bes[i*HID + n]; m = ms[i*HID + n];
        v  = vs [i*HID + n]; b1 = b1s[i*HID + n];
    }
    float s = g * rsqrtf(v + 1e-5f);
    bns[tid] = s;
    bnt[tid] = (b1 - m) * s + be;
}

__global__ void fill0_kernel(float* __restrict__ p, long n) {
    long i = (long)blockIdx.x * blockDim.x + threadIdx.x;
    if (i < n) p[i] = 0.f;
}

// ---------------------------------------------------------------------------
// Edge scatter-add: one wave per edge, float4 per lane, 4 global f32 atomics.
// x and agg are L2-resident (25.6MB), so gathers hit L2.
// ---------------------------------------------------------------------------
__global__ __launch_bounds__(256) void agg_kernel(
    const float* __restrict__ x, const int* __restrict__ src,
    const int* __restrict__ dst, float* __restrict__ agg, int n_edges)
{
    int e = blockIdx.x * 8 + (threadIdx.x >> 5);
    if (e >= n_edges) return;
    int lane = threadIdx.x & 31;
    long s = src[e], d = dst[e];
    float4 v = ((const float4*)(x + s * HID))[lane];
    float* p = agg + d * HID + (long)lane * 4;
    atomicAdd(p + 0, v.x); atomicAdd(p + 1, v.y);
    atomicAdd(p + 2, v.z); atomicAdd(p + 3, v.w);
}

// ---------------------------------------------------------------------------
// Fused GIN MLP for 64 nodes / block (8 wave32s):
//   stage (x+agg) -> LDS bf16 hi/lo
//   GEMM1 (WMMA bf16 split) -> BN+ReLU -> LDS bf16 hi/lo
//   GEMM2 (WMMA bf16 split) -> +b2, ReLU -> global out + pooled column sums
// Each wave: rtile = wave&3 (16 rows), 4 col-tiles (cgroup = wave>>2).
// 96 static v_wmma_f32_16x16x32_bf16 per wave.
// ---------------------------------------------------------------------------
__global__ __launch_bounds__(256) void mlp_kernel(
    const float*  __restrict__ xin, const float* __restrict__ agg,
    const __bf16* __restrict__ w1hi, const __bf16* __restrict__ w1lo,
    const __bf16* __restrict__ w2hi, const __bf16* __restrict__ w2lo,
    const float*  __restrict__ bns,  const float* __restrict__ bnt,
    const float*  __restrict__ b2,   float* __restrict__ out,
    float* __restrict__ pooled, int n_nodes)
{
    __shared__ __align__(32) __bf16 sh_hi[TM * HID];
    __shared__ __align__(32) __bf16 sh_lo[TM * HID];
    __shared__ float pool[HID];

    const int tid    = threadIdx.x;
    const int wave   = tid >> 5;
    const int lane   = tid & 31;
    const int laneHi = (lane >> 4) & 1;
    const int ln     = lane & 15;
    const int rtile  = wave & 3;
    const int cgrp   = wave >> 2;
    const long node0 = (long)blockIdx.x * TM;

    if (tid < HID) pool[tid] = 0.f;

    // ---- stage s = x + agg as bf16 hi/lo in LDS (zero-pad tail rows) ----
    #pragma unroll
    for (int it = 0; it < 8; ++it) {
        int f4   = tid + 256 * it;          // 2048 float4 slots = 64x128
        int row  = f4 >> 5;
        int col4 = (f4 & 31) * 4;
        long node = node0 + row;
        float4 v = make_float4(0.f, 0.f, 0.f, 0.f);
        if (node < n_nodes) {
            float4 a = *(const float4*)(xin + node * HID + col4);
            float4 b = *(const float4*)(agg + node * HID + col4);
            v = make_float4(a.x + b.x, a.y + b.y, a.z + b.z, a.w + b.w);
        }
        int base = row * HID + col4;
        float vv[4] = {v.x, v.y, v.z, v.w};
        #pragma unroll
        for (int q = 0; q < 4; ++q) {
            __bf16 h = (__bf16)vv[q];
            sh_hi[base + q] = h;
            sh_lo[base + q] = (__bf16)(vv[q] - (float)h);
        }
    }
    __syncthreads();

    const int arow = rtile * 16 + ln;

    // ---- GEMM1: h1 = s @ W1^T ----
    f32x8 acc[4] = {};
    #pragma unroll
    for (int kt = 0; kt < 4; ++kt) {
        int kb    = kt * 32 + laneHi * 16;   // A fragment base
        int khalf = kt * 32 + laneHi * 16;   // B fragment base (contiguous 16)
        bf16x16 ahi = load_a(&sh_hi[arow * HID], kb);
        bf16x16 alo = load_a(&sh_lo[arow * HID], kb);
        #pragma unroll
        for (int i = 0; i < 4; ++i) {
            int n = (cgrp * 4 + i) * 16 + ln;
            bf16x16 bhi = *(const bf16x16*)(w1hi + (size_t)n * HID + khalf);
            bf16x16 blo = *(const bf16x16*)(w1lo + (size_t)n * HID + khalf);
            acc[i] = wmma_bf16(ahi, bhi, acc[i]);
            acc[i] = wmma_bf16(ahi, blo, acc[i]);
            acc[i] = wmma_bf16(alo, bhi, acc[i]);
        }
    }
    __syncthreads();   // done reading input LDS

    // ---- BN + ReLU, re-split mid activations into LDS ----
    #pragma unroll
    for (int i = 0; i < 4; ++i) {
        int n = (cgrp * 4 + i) * 16 + ln;
        float s = bns[n], t = bnt[n];
        #pragma unroll
        for (int r = 0; r < 8; ++r) {
            int m = rtile * 16 + laneHi * 8 + r;
            float v = fmaxf(acc[i][r] * s + t, 0.f);
            __bf16 h = (__bf16)v;
            sh_hi[m * HID + n] = h;
            sh_lo[m * HID + n] = (__bf16)(v - (float)h);
        }
    }
    __syncthreads();

    // ---- GEMM2: h2 = relu(mid @ W2^T + b2) ----
    f32x8 acc2[4] = {};
    #pragma unroll
    for (int kt = 0; kt < 4; ++kt) {
        int kb    = kt * 32 + laneHi * 16;
        int khalf = kt * 32 + laneHi * 16;
        bf16x16 ahi = load_a(&sh_hi[arow * HID], kb);
        bf16x16 alo = load_a(&sh_lo[arow * HID], kb);
        #pragma unroll
        for (int i = 0; i < 4; ++i) {
            int n = (cgrp * 4 + i) * 16 + ln;
            bf16x16 bhi = *(const bf16x16*)(w2hi + (size_t)n * HID + khalf);
            bf16x16 blo = *(const bf16x16*)(w2lo + (size_t)n * HID + khalf);
            acc2[i] = wmma_bf16(ahi, bhi, acc2[i]);
            acc2[i] = wmma_bf16(ahi, blo, acc2[i]);
            acc2[i] = wmma_bf16(alo, bhi, acc2[i]);
        }
    }

    // ---- epilogue: store layer output, accumulate pooled column sums ----
    #pragma unroll
    for (int i = 0; i < 4; ++i) {
        int n = (cgrp * 4 + i) * 16 + ln;
        float b = b2[n];
        float colsum = 0.f;
        #pragma unroll
        for (int r = 0; r < 8; ++r) {
            int m = rtile * 16 + laneHi * 8 + r;
            long node = node0 + m;
            float v = fmaxf(acc2[i][r] + b, 0.f);
            if (node < n_nodes) {
                out[node * HID + n] = v;
                colsum += v;
            }
        }
        atomicAdd(&pool[n], colsum);
    }
    __syncthreads();
    if (tid < HID) atomicAdd(&pooled[tid], pool[tid]);
}

// ---------------------------------------------------------------------------
// Readout: out[c] = bll[c] + sum_j pooled[j] * Wll[c][j]   (10 x 512)
// ---------------------------------------------------------------------------
__global__ __launch_bounds__(256) void readout_kernel(
    const float* __restrict__ pooled, const float* __restrict__ Wll,
    const float* __restrict__ bll, float* __restrict__ outp)
{
    __shared__ float red[256];
    int tid = threadIdx.x;
    int c   = blockIdx.x;                 // 10 classes
    float p = pooled[tid]       * Wll[(size_t)c * 512 + tid]
            + pooled[tid + 256] * Wll[(size_t)c * 512 + tid + 256];
    red[tid] = p;
    __syncthreads();
    for (int s = 128; s > 0; s >>= 1) {
        if (tid < s) red[tid] += red[tid + s];
        __syncthreads();
    }
    if (tid == 0) outp[c] = red[0] + bll[c];
}

// ---------------------------------------------------------------------------
extern "C" void kernel_launch(void* const* d_in, const int* in_sizes, int n_in,
                              void* d_out, int out_size, void* d_ws, size_t ws_size,
                              hipStream_t stream)
{
    const float* x     = (const float*)d_in[0];
    const int*   ei    = (const int*)  d_in[1];
    const float* W1_0  = (const float*)d_in[2];
    const float* b1_0  = (const float*)d_in[3];
    const float* g0    = (const float*)d_in[4];
    const float* be0   = (const float*)d_in[5];
    const float* m0    = (const float*)d_in[6];
    const float* v0    = (const float*)d_in[7];
    const float* W2_0  = (const float*)d_in[8];
    const float* b2_0  = (const float*)d_in[9];
    const float* W1s   = (const float*)d_in[10];
    const float* b1s   = (const float*)d_in[11];
    const float* gs    = (const float*)d_in[12];
    const float* bes   = (const float*)d_in[13];
    const float* ms    = (const float*)d_in[14];
    const float* vs    = (const float*)d_in[15];
    const float* W2s   = (const float*)d_in[16];
    const float* b2s   = (const float*)d_in[17];
    const float* Wll   = (const float*)d_in[18];
    const float* bll   = (const float*)d_in[19];

    const int  N  = in_sizes[0] / HID;     // 50000 nodes
    const int  E  = in_sizes[1] / 2;       // 600000 edges
    const long Nf = (long)N * HID;

    const int* src = ei;
    const int* dst = ei + E;

    // ---- workspace carve-up (256B aligned regions) ----
    char* wsb = (char*)d_ws;
    size_t off = 0;
    auto sub = [&](size_t bytes) -> void* {
        void* r = wsb + off;
        off = (off + bytes + 255) & ~(size_t)255;
        return r;
    };
    float*  agg    = (float*) sub((size_t)Nf * 4);
    float*  hA     = (float*) sub((size_t)Nf * 4);
    float*  hB     = (float*) sub((size_t)Nf * 4);
    __bf16* whi    = (__bf16*)sub((size_t)NMAT * MATSZ * 2);
    __bf16* wlo    = (__bf16*)sub((size_t)NMAT * MATSZ * 2);
    float*  bns    = (float*) sub(512 * 4);
    float*  bnt    = (float*) sub(512 * 4);
    float*  pooled = (float*) sub(512 * 4);
    (void)ws_size; (void)n_in; (void)out_size;

    // ---- one-time (per launch) prep ----
    pack_kernel<<<dim3(MATSZ / 256, NMAT), 256, 0, stream>>>(
        W1_0, W2_0, W1s, W2s, whi, wlo);
    bnprep_kernel<<<1, 512, 0, stream>>>(
        g0, be0, m0, v0, b1_0, gs, bes, ms, vs, b1s, bns, bnt);
    fill0_kernel<<<(512 + 255) / 256, 256, 0, stream>>>(pooled, 512);

    // ---- 4 GIN layers ----
    const float* cur = x;
    for (int l = 0; l < 4; ++l) {
        fill0_kernel<<<(int)((Nf + 255) / 256), 256, 0, stream>>>(agg, Nf);
        agg_kernel<<<(E + 7) / 8, 256, 0, stream>>>(cur, src, dst, agg, E);

        const __bf16* w1h = whi + (size_t)(2 * l)     * MATSZ;
        const __bf16* w1l = wlo + (size_t)(2 * l)     * MATSZ;
        const __bf16* w2h = whi + (size_t)(2 * l + 1) * MATSZ;
        const __bf16* w2l = wlo + (size_t)(2 * l + 1) * MATSZ;
        const float*  b2p = (l == 0) ? b2_0 : (b2s + (size_t)(l - 1) * HID);
        float* outb = (l & 1) ? hB : hA;   // l0->A, l1->B, l2->A, l3->B

        mlp_kernel<<<(N + TM - 1) / TM, 256, 0, stream>>>(
            cur, agg, w1h, w1l, w2h, w2l,
            bns + l * HID, bnt + l * HID, b2p,
            outb, pooled + l * HID, N);

        cur = outb;
    }

    // ---- final readout ----
    readout_kernel<<<10, 256, 0, stream>>>(pooled, Wll, bll, (float*)d_out);
}
